// RNN_29858612641988
// MI455X (gfx1250) — matrix operand order repaired and checked by reading, not compiled, not run on hardware
//
#include <hip/hip_runtime.h>
#include <hip/hip_bf16.h>

// ---------------------------------------------------------------------------
// 5-layer LSTM (B=2048,T=512,IN=3,H=64) + FC head for MI455X (gfx1250).
// Recurrent GEMM per step done with v_wmma_f32_16x16x32_bf16, weights held in
// VGPR fragments, gates fanned in through LDS. One workgroup = 16 batch rows,
// 8 waves (wave32), T-loop inside the kernel. Branchless activations using a
// single fused v_exp per gate (tanh(x) = 2*sigmoid(2x)-1, scale/log2e folded
// into one constant); cell state c lives in per-thread VGPRs.
// Workspace: one bf16 [B,T,H] sequence buffer = 128 MiB, used in place.
// ---------------------------------------------------------------------------

constexpr int Bn = 2048;
constexpr int Tn = 512;
constexpr int Hn = 64;     // hidden
constexpr int Gn = 4 * Hn; // 256 gate columns

typedef __attribute__((ext_vector_type(16))) __bf16 v16bf;
typedef __attribute__((ext_vector_type(8)))  __bf16 bf16x8;
typedef __attribute__((ext_vector_type(4)))  __bf16 bf16x4;
typedef __attribute__((ext_vector_type(8)))  float  v8f;

#if __has_builtin(__builtin_amdgcn_exp2f)
static __device__ __forceinline__ float exp2_raw(float x) {
    return __builtin_amdgcn_exp2f(x);          // raw v_exp_f32
}
#else
static __device__ __forceinline__ float exp2_raw(float x) {
    return __expf(x * 0.69314718056f);
}
#endif

constexpr float LOG2E = 1.44269504089f;

static __device__ __forceinline__ __bf16 f2bf(float f) {
    unsigned int u = __builtin_bit_cast(unsigned int, f);
    u = u + 0x7FFFu + ((u >> 16) & 1u);          // round-to-nearest-even
    unsigned short s = (unsigned short)(u >> 16);
    return __builtin_bit_cast(__bf16, s);
}
static __device__ __forceinline__ float bf2f(__bf16 b) {
    unsigned int u = ((unsigned int)__builtin_bit_cast(unsigned short, b)) << 16;
    return __builtin_bit_cast(float, u);
}

// Branchless gate activation: sigmoid(x), or tanh(x) = 2*sigmoid(2x)-1.
// One v_mul + v_exp + v_add + v_rcp + v_fma; selection via cndmask only.
static __device__ __forceinline__ float gate_act(float x, float c0,
                                                 float mul, float add) {
    const float e = exp2_raw(x * c0);                   // exp(-scale*x)
    const float s = __builtin_amdgcn_rcpf(1.0f + e);    // e->inf => s->0 (safe)
    return fmaf(mul, s, add);
}
static __device__ __forceinline__ float tanh_fast(float x) {
    const float e = exp2_raw(x * (-2.0f * LOG2E));
    const float s = __builtin_amdgcn_rcpf(1.0f + e);
    return fmaf(2.0f, s, -1.0f);
}

// ---------------------------------------------------------------------------
// One LSTM layer. FIRST: input is fp32 [B,T,3] padded to KIN=32 columns.
// Otherwise: input is bf16 [B,T,64] (previous layer's output, in place).
// ---------------------------------------------------------------------------
template <bool FIRST, int KIN>
__global__ __launch_bounds__(256)
void lstm_layer_kernel(const float* __restrict__ x0,
                       const __bf16* __restrict__ xin,
                       const float* __restrict__ w_ih,   // [256, FIRST?3:64]
                       const float* __restrict__ w_hh,   // [256, 64]
                       const float* __restrict__ b_ih,   // [256]
                       const float* __restrict__ b_hh,   // [256]
                       __bf16* __restrict__ out_seq)     // [B,T,64] bf16
{
    constexpr int KTOT = KIN + Hn;     // 96 (layer0) or 128
    constexpr int NKS  = KTOT / 32;    // 3 or 4 WMMA k-steps

    __shared__ __align__(16) __bf16 sA[16][KTOT]; // A = [x | h], bf16
    __shared__ float sG[16][Gn];                  // activated gates

    const int tid  = threadIdx.x;
    const int wave = tid >> 5;
    const int lane = tid & 31;
    const int b0   = blockIdx.x * 16;

    const int n     = lane & 15;
    const int khalf = lane >> 4;

    // ---- Preload B fragments (weights) for this wave's 2 N-tiles ----------
    // B k order chosen here (contiguous, not AMD's table order): lanes 0-15 hold K=0..15, lanes 16-31
    // hold K=16..31, column = lane&15; A fragment uses the same k order.
    v16bf bfrag[2][NKS];
    float bias[2];
#pragma unroll
    for (int tt = 0; tt < 2; ++tt) {
        const int tile = wave * 2 + tt;
        const int col  = tile * 16 + n;
        bias[tt] = b_ih[col] + b_hh[col];
        const int kb0 = khalf * 16;
#pragma unroll
        for (int s = 0; s < NKS; ++s) {
            v16bf f;
#pragma unroll
            for (int e = 0; e < 16; ++e) {
                const int k = s * 32 + kb0 + e;
                float w;
                if (FIRST)
                    w = (k < KIN) ? ((k < 3) ? w_ih[col * 3 + k] : 0.0f)
                                  : w_hh[col * Hn + (k - KIN)];
                else
                    w = (k < KIN) ? w_ih[col * Hn + k]
                                  : w_hh[col * Hn + (k - KIN)];
                f[e] = f2bf(w);
            }
            bfrag[tt][s] = f;
        }
    }

    // Activation constants, wave-uniform (tiles (2w,2w+1) -> gate = w>>1;
    // gate 2 ("g") = tanh, i/f/o = sigmoid). Pure cndmask selection.
    const bool  is_g = ((wave >> 1) == 2);
    const float actc = is_g ? (-2.0f * LOG2E) : (-LOG2E);
    const float actm = is_g ? 2.0f : 1.0f;
    const float acta = is_g ? -1.0f : 0.0f;

    // Cell state: fixed tid -> (m, hc) mapping every step => keep in VGPRs.
    const int um  = tid >> 4;          // update-phase row
    const int uhc = (tid & 15) * 4;    // update-phase first h column
    float creg[4] = {0.0f, 0.0f, 0.0f, 0.0f};

    // ---- Zero-init the h region of A ---------------------------------------
    for (int idx = tid; idx < 16 * Hn; idx += 256) {
        sA[idx >> 6][KIN + (idx & 63)] = f2bf(0.0f);
    }
    __syncthreads();

    for (int t = 0; t < Tn; ++t) {
        // ---- Stage x_t tile into sA[:, 0:KIN] ------------------------------
        if (FIRST) {
            for (int idx = tid; idx < 16 * KIN; idx += 256) {
                const int m = idx / KIN, k = idx % KIN;
                const float v = (k < 3)
                    ? x0[((size_t)(b0 + m) * Tn + t) * 3 + k] : 0.0f;
                sA[m][k] = f2bf(v);
            }
        } else {
            const __bf16* p = &xin[((size_t)(b0 + um) * Tn + t) * Hn + uhc];
            *(bf16x4*)&sA[um][uhc] = *(const bf16x4*)p;
            if (t + 1 < Tn) __builtin_prefetch(p + Hn, 0, 0); // next timestep
        }
        __syncthreads();

        // ---- gates = [x|h] @ [w_ih|w_hh]^T + bias via WMMA -----------------
        {
            v8f acc0, acc1;
#pragma unroll
            for (int r = 0; r < 8; ++r) { acc0[r] = bias[0]; acc1[r] = bias[1]; }

#pragma unroll
            for (int s = 0; s < NKS; ++s) {
                // A fragment (16-bit A 16x32 layout): lane m = lane&15,
                // contiguous 16-elem run at K = s*32 + khalf*16 (matches B).
                const bf16x8 lo = *(const bf16x8*)&sA[n][s * 32 + khalf * 16];
                const bf16x8 hi = *(const bf16x8*)&sA[n][s * 32 + khalf * 16 + 8];
                v16bf a;
#pragma unroll
                for (int e = 0; e < 8; ++e) { a[e] = lo[e]; a[8 + e] = hi[e]; }

                acc0 = __builtin_amdgcn_wmma_f32_16x16x32_bf16(
                    false, a, false, bfrag[0][s], (short)0, acc0, false, false);
                acc1 = __builtin_amdgcn_wmma_f32_16x16x32_bf16(
                    false, a, false, bfrag[1][s], (short)0, acc1, false, false);
            }

            const int mhi = khalf * 8;
#pragma unroll
            for (int r = 0; r < 8; ++r) {
                sG[mhi + r][(wave * 2 + 0) * 16 + n] =
                    gate_act(acc0[r], actc, actm, acta);
                sG[mhi + r][(wave * 2 + 1) * 16 + n] =
                    gate_act(acc1[r], actc, actm, acta);
            }
        }
        __syncthreads();

        // ---- c,h update; h feeds sA (next step) and out_seq ----------------
        {
            bf16x4 hv;
#pragma unroll
            for (int e = 0; e < 4; ++e) {
                const int hc = uhc + e;
                const float i = sG[um][hc];
                const float f = sG[um][Hn + hc];
                const float g = sG[um][2 * Hn + hc];
                const float o = sG[um][3 * Hn + hc];
                const float c = f * creg[e] + i * g;
                creg[e] = c;
                const float h = o * tanh_fast(c);
                const __bf16 hb = f2bf(h);
                sA[um][KIN + hc] = hb;
                hv[e] = hb;
            }
            *(bf16x4*)&out_seq[((size_t)(b0 + um) * Tn + t) * Hn + uhc] = hv;
        }
        __syncthreads();
    }
}

// ---------------------------------------------------------------------------
// FC head: out = relu(h_T @ fc1^T + b1) @ fc2^T + b2, h_T = seq[:, T-1, :]
// ---------------------------------------------------------------------------
__global__ __launch_bounds__(256)
void fc_head_kernel(const __bf16* __restrict__ seq,
                    const float* __restrict__ fc1_w, const float* __restrict__ fc1_b,
                    const float* __restrict__ fc2_w, const float* __restrict__ fc2_b,
                    float* __restrict__ out)
{
    const int b = blockIdx.x * blockDim.x + threadIdx.x;
    if (b >= Bn) return;

    float h[Hn];
    const __bf16* hp = &seq[((size_t)b * Tn + (Tn - 1)) * Hn];
#pragma unroll
    for (int k = 0; k < Hn; ++k) h[k] = bf2f(hp[k]);

    float acc2 = fc2_b[0];
    for (int j = 0; j < 50; ++j) {
        float a = fc1_b[j];
#pragma unroll
        for (int k = 0; k < Hn; ++k) a += h[k] * fc1_w[j * Hn + k];
        a = fmaxf(a, 0.0f);
        acc2 += a * fc2_w[j];
    }
    out[b] = acc2;
}

// ---------------------------------------------------------------------------
extern "C" void kernel_launch(void* const* d_in, const int* in_sizes, int n_in,
                              void* d_out, int out_size, void* d_ws, size_t ws_size,
                              hipStream_t stream)
{
    const float* x      = (const float*)d_in[0];
    const float* w_ih0  = (const float*)d_in[1];
    const float* w_ihr  = (const float*)d_in[2];  // (4, 256, 64)
    const float* w_hh   = (const float*)d_in[3];  // (5, 256, 64)
    const float* b_ih   = (const float*)d_in[4];  // (5, 256)
    const float* b_hh   = (const float*)d_in[5];  // (5, 256)
    const float* fc1_w  = (const float*)d_in[6];
    const float* fc1_b  = (const float*)d_in[7];
    const float* fc2_w  = (const float*)d_in[8];
    const float* fc2_b  = (const float*)d_in[9];
    float* out = (float*)d_out;

    // bf16 [B,T,H] sequence buffer, used in place across layers: 128 MiB.
    __bf16* seq = (__bf16*)d_ws;

    const dim3 grid(Bn / 16), block(256);

    lstm_layer_kernel<true, 32><<<grid, block, 0, stream>>>(
        x, nullptr, w_ih0, w_hh, b_ih, b_hh, seq);

    for (int l = 1; l < 5; ++l) {
        lstm_layer_kernel<false, 64><<<grid, block, 0, stream>>>(
            nullptr, seq,
            w_ihr + (size_t)(l - 1) * Gn * Hn,
            w_hh  + (size_t)l * Gn * Hn,
            b_ih  + l * Gn,
            b_hh  + l * Gn,
            seq);
    }

    fc_head_kernel<<<dim3(Bn / 256), block, 0, stream>>>(
        seq, fc1_w, fc1_b, fc2_w, fc2_b, out);
}
